// AttentionBlock_32908039421954
// MI455X (gfx1250) — compile-verified, matrix operand order checked by reading, not run on hardware
//
#include <hip/hip_runtime.h>
#include <math.h>

// ---------------------------------------------------------------------------
// Graph attention block for MI455X (gfx1250, wave32, WMMA).
// Pipeline: bf16 casts -> WMMA QKV GEMM -> edge scores + segment softmax
//           (atomics) -> weighted scatter -> WMMA output GEMM.
// GEMM: 2x4 register blocking per wave (8 WMMAs per k-step, 12 b128 loads)
//       with explicit next-k prefetch registers for deep pipelining.
// ---------------------------------------------------------------------------

typedef __bf16 bf16_t;
typedef __attribute__((ext_vector_type(16))) __bf16 v16bf;
typedef __attribute__((ext_vector_type(8)))  float  v8f;

static constexpr int HEADS    = 8;
static constexpr int DHEAD    = 64;    // dK == dV
static constexpr int D_EMB    = 512;
static constexpr int QKV_COLS = 1536;  // (2*dK + dV) * H

// ---------------------------------------------------------------------------
// Fragment load: 16-bit A/B fragment per CDNA5 ISA layout.
// Lane l holds row/col (l & 15); half = l>>4 selects K sub-range.
// Vector elems 0..7  = K: base..base+7       (VGPR0..3, lo/hi pairs)
// Vector elems 8..15 = K: 16+base..16+base+7 (VGPR4..7),  base = 8*half.
// ---------------------------------------------------------------------------
__device__ __forceinline__ v16bf load_frag16(const bf16_t* p) {
    union { uint4 u[2]; v16bf v; } r;
    r.u[0] = *reinterpret_cast<const uint4*>(p);
    r.u[1] = *reinterpret_cast<const uint4*>(p + 16);
    return r.v;
}

__device__ __forceinline__ v8f wmma_bf16(v16bf a, v16bf b, v8f c) {
    return __builtin_amdgcn_wmma_f32_16x16x32_bf16(
               false, a, false, b, (short)0, c, false, false);
}

// C = A[M,K](bf16, row-major) * B[Ncols,K](bf16, row-major)^T + bias[Ncols]
// One wave -> 32(M) x 64(N): 2x4 accumulators, A frags reused 4x, B frags 2x.
// Block = 256 threads = 8 waves -> 32 x 512 block tile.
__global__ __launch_bounds__(256)
void wmma_gemm_bias(const bf16_t* __restrict__ A, const bf16_t* __restrict__ B,
                    const float* __restrict__ bias, float* __restrict__ C,
                    int M, int Ncols, int K) {
    const int lane = threadIdx.x & 31;
    const int wave = threadIdx.x >> 5;
    const int m0 = blockIdx.x * 32;
    const int n0 = blockIdx.y * 512 + wave * 64;
    if (n0 >= Ncols) return;                 // wave-uniform

    const int half = lane >> 4;              // 0 or 1
    const int mn   = lane & 15;              // row (A) / col (B) within tile

    int ar0 = m0 + mn;      if (ar0 >= M) ar0 = M - 1;   // clamp (branchless)
    int ar1 = m0 + 16 + mn; if (ar1 >= M) ar1 = M - 1;
    const bf16_t* Ap0 = A + (size_t)ar0 * K + half * 8;
    const bf16_t* Ap1 = A + (size_t)ar1 * K + half * 8;
    const bf16_t* Bp  = B + (size_t)(n0 + mn) * K + half * 8;
    const size_t  bstep = (size_t)16 * K;    // 16 B-rows (= 16 C columns)

    v8f acc[2][4] = {};

    // prime the pipeline: fragments for k0 = 0
    v16bf a0 = load_frag16(Ap0);
    v16bf a1 = load_frag16(Ap1);
    v16bf b0 = load_frag16(Bp);
    v16bf b1 = load_frag16(Bp + bstep);
    v16bf b2 = load_frag16(Bp + 2 * bstep);
    v16bf b3 = load_frag16(Bp + 3 * bstep);

    #pragma unroll
    for (int k0 = 32; k0 < 512; k0 += 32) {  // K == 512 for both GEMMs
        // issue next-step loads before consuming current fragments
        v16bf a0n = load_frag16(Ap0 + k0);
        v16bf a1n = load_frag16(Ap1 + k0);
        v16bf b0n = load_frag16(Bp + k0);
        v16bf b1n = load_frag16(Bp + bstep + k0);
        v16bf b2n = load_frag16(Bp + 2 * bstep + k0);
        v16bf b3n = load_frag16(Bp + 3 * bstep + k0);

        acc[0][0] = wmma_bf16(a0, b0, acc[0][0]);
        acc[0][1] = wmma_bf16(a0, b1, acc[0][1]);
        acc[0][2] = wmma_bf16(a0, b2, acc[0][2]);
        acc[0][3] = wmma_bf16(a0, b3, acc[0][3]);
        acc[1][0] = wmma_bf16(a1, b0, acc[1][0]);
        acc[1][1] = wmma_bf16(a1, b1, acc[1][1]);
        acc[1][2] = wmma_bf16(a1, b2, acc[1][2]);
        acc[1][3] = wmma_bf16(a1, b3, acc[1][3]);

        a0 = a0n; a1 = a1n; b0 = b0n; b1 = b1n; b2 = b2n; b3 = b3n;
    }
    // drain last k-step
    acc[0][0] = wmma_bf16(a0, b0, acc[0][0]);
    acc[0][1] = wmma_bf16(a0, b1, acc[0][1]);
    acc[0][2] = wmma_bf16(a0, b2, acc[0][2]);
    acc[0][3] = wmma_bf16(a0, b3, acc[0][3]);
    acc[1][0] = wmma_bf16(a1, b0, acc[1][0]);
    acc[1][1] = wmma_bf16(a1, b1, acc[1][1]);
    acc[1][2] = wmma_bf16(a1, b2, acc[1][2]);
    acc[1][3] = wmma_bf16(a1, b3, acc[1][3]);

    // Store: C layout -> VGPR v holds M = v + 8*half, N = lane&15.
    #pragma unroll
    for (int t = 0; t < 4; ++t) {
        const int col = n0 + t * 16 + mn;
        const float bv = bias ? bias[col] : 0.0f;
        #pragma unroll
        for (int rg = 0; rg < 2; ++rg) {
            #pragma unroll
            for (int v = 0; v < 8; ++v) {
                const int row = m0 + rg * 16 + half * 8 + v;
                if (row < M)
                    C[(size_t)row * Ncols + col] = acc[rg][t][v] + bv;
            }
        }
    }
}

// ---------------------------------------------------------------------------
// float -> bf16 cast
// ---------------------------------------------------------------------------
__global__ void cast_to_bf16(const float* __restrict__ in,
                             bf16_t* __restrict__ out, size_t n) {
    size_t i = (size_t)blockIdx.x * blockDim.x + threadIdx.x;
    const size_t stride = (size_t)gridDim.x * blockDim.x;
    for (; i < n; i += stride) out[i] = (bf16_t)in[i];
}

// ---------------------------------------------------------------------------
// Ordered-uint encoding for float atomic max (monotone; key 0 < -FLT_MAX).
// ---------------------------------------------------------------------------
__device__ __forceinline__ unsigned int float_to_ordered(float f) {
    unsigned int u = __float_as_uint(f);
    return (u & 0x80000000u) ? ~u : (u | 0x80000000u);
}
__device__ __forceinline__ float ordered_to_float(unsigned int k) {
    return (k & 0x80000000u) ? __uint_as_float(k ^ 0x80000000u)
                             : __uint_as_float(~k);
}

// ---------------------------------------------------------------------------
// Per-edge attention scores + segment-max.
// One wave per edge. Lane -> (head h = lane>>2, quarter q = lane&3);
// each lane dots 16 contiguous dims via 4 float4 loads, then a 2-step
// butterfly over the 4 lanes of each head. All 8 heads reduce in parallel.
// ---------------------------------------------------------------------------
__global__ __launch_bounds__(256)
void edge_scores(const float* __restrict__ qkv, const int* __restrict__ send,
                 const int* __restrict__ recv, float* __restrict__ scores,
                 unsigned int* __restrict__ smaxkey, int E) {
    const int lane = threadIdx.x & 31;
    const int e = blockIdx.x * 8 + (threadIdx.x >> 5);
    if (e >= E) return;
    const int s = send[e], r = recv[e];
    const int h = lane >> 2;                 // head 0..7
    const int q = lane & 3;                  // quarter 0..3
    const float* Qr = qkv + (size_t)r * QKV_COLS + h * DHEAD + q * 16;
    const float* Ks = qkv + (size_t)s * QKV_COLS + D_EMB + h * DHEAD + q * 16;

    float p = 0.0f;
    #pragma unroll
    for (int j = 0; j < 4; ++j) {
        const float4 qv = *reinterpret_cast<const float4*>(Qr + j * 4);
        const float4 kv = *reinterpret_cast<const float4*>(Ks + j * 4);
        p += qv.x * kv.x + qv.y * kv.y + qv.z * kv.z + qv.w * kv.w;
    }
    p += __shfl_xor(p, 1, 32);
    p += __shfl_xor(p, 2, 32);

    if (q == 0) {
        const float sc = p * 0.125f;         // 1/sqrt(64)
        scores[(size_t)e * HEADS + h] = sc;
        atomicMax(&smaxkey[(size_t)r * HEADS + h], float_to_ordered(sc));
    }
}

// ---------------------------------------------------------------------------
// ex = exp(score - smax[recv]);  denom[recv] += ex   (in-place on scores)
// ---------------------------------------------------------------------------
__global__ void edge_exp(float* __restrict__ scores, const int* __restrict__ recv,
                         const unsigned int* __restrict__ smaxkey,
                         float* __restrict__ denom, size_t total) {
    const size_t i = (size_t)blockIdx.x * blockDim.x + threadIdx.x;
    if (i >= total) return;
    const size_t e = i >> 3;
    const int   h = (int)(i & 7);
    const int   r = recv[e];
    const float mx = ordered_to_float(smaxkey[(size_t)r * HEADS + h]);
    const float ex = expf(scores[i] - mx);
    scores[i] = ex;
    atomicAdd(&denom[(size_t)r * HEADS + h], ex);
}

// ---------------------------------------------------------------------------
// msg[recv, h, :] += (ex / denom[recv,h]) * V[send, h, :]
// One wave per edge; lanes cover the 64-dim V in two halves.
// ---------------------------------------------------------------------------
__global__ __launch_bounds__(256)
void edge_scatter(const float* __restrict__ qkv, const float* __restrict__ ex,
                  const float* __restrict__ denom, const int* __restrict__ send,
                  const int* __restrict__ recv, float* __restrict__ msg, int E) {
    const int lane = threadIdx.x & 31;
    const int e = blockIdx.x * 8 + (threadIdx.x >> 5);
    if (e >= E) return;
    const int s = send[e], r = recv[e];
    const float* Vs = qkv + (size_t)s * QKV_COLS + 2 * D_EMB;  // V block
    float* Mr = msg + (size_t)r * (HEADS * DHEAD);

    #pragma unroll
    for (int h = 0; h < HEADS; ++h) {
        const float w = ex[(size_t)e * HEADS + h] / denom[(size_t)r * HEADS + h];
        atomicAdd(&Mr[h * DHEAD + lane],      w * Vs[h * DHEAD + lane]);
        atomicAdd(&Mr[h * DHEAD + 32 + lane], w * Vs[h * DHEAD + 32 + lane]);
    }
}

// ---------------------------------------------------------------------------
// Host launcher
// ---------------------------------------------------------------------------
static inline size_t align256(size_t x) { return (x + 255) & ~(size_t)255; }

extern "C" void kernel_launch(void* const* d_in, const int* in_sizes, int n_in,
                              void* d_out, int out_size, void* d_ws, size_t ws_size,
                              hipStream_t stream) {
    const float* x      = (const float*)d_in[0];
    const int*   edge   = (const int*)d_in[1];
    const float* W_qkv  = (const float*)d_in[2];
    const float* b_qkv  = (const float*)d_in[3];
    const float* W_ff   = (const float*)d_in[4];
    const float* b_ff   = (const float*)d_in[5];

    const int Nn = in_sizes[0] / D_EMB;       // 50000
    const int E  = in_sizes[1] / 2;           // 800000
    const int* send = edge;
    const int* recv = edge + E;

    // scratch layout
    char* ws = (char*)d_ws;
    auto carve = [&](size_t bytes) { char* p = ws; ws += align256(bytes); return p; };
    float*        qkv     = (float*)       carve((size_t)Nn * QKV_COLS * sizeof(float));
    float*        msg     = (float*)       carve((size_t)Nn * D_EMB * sizeof(float));
    bf16_t*       xb      = (bf16_t*)      carve((size_t)Nn * D_EMB * sizeof(bf16_t));
    bf16_t*       msgb    = (bf16_t*)      carve((size_t)Nn * D_EMB * sizeof(bf16_t));
    bf16_t*       wqkvb   = (bf16_t*)      carve((size_t)QKV_COLS * D_EMB * sizeof(bf16_t));
    bf16_t*       wffb    = (bf16_t*)      carve((size_t)D_EMB * D_EMB * sizeof(bf16_t));
    float*        scores  = (float*)       carve((size_t)E * HEADS * sizeof(float));
    unsigned int* smaxkey = (unsigned int*)carve((size_t)Nn * HEADS * sizeof(unsigned int));
    float*        denom   = (float*)       carve((size_t)Nn * HEADS * sizeof(float));

    // 1) casts to bf16
    {
        size_t n = (size_t)Nn * D_EMB;
        cast_to_bf16<<<2048, 256, 0, stream>>>(x, xb, n);
        cast_to_bf16<<<1024, 256, 0, stream>>>(W_qkv, wqkvb, (size_t)QKV_COLS * D_EMB);
        cast_to_bf16<<<512,  256, 0, stream>>>(W_ff,  wffb,  (size_t)D_EMB * D_EMB);
    }

    // 2) zero / init accumulators (ordered key 0 encodes < -FLT_MAX)
    hipMemsetAsync(msg,     0, (size_t)Nn * D_EMB * sizeof(float), stream);
    hipMemsetAsync(smaxkey, 0, (size_t)Nn * HEADS * sizeof(unsigned int), stream);
    hipMemsetAsync(denom,   0, (size_t)Nn * HEADS * sizeof(float), stream);

    // 3) QKV projection: qkv = x @ W_qkv^T + b_qkv   [Nn x 1536]
    {
        dim3 grid((Nn + 31) / 32, QKV_COLS / 512);
        wmma_gemm_bias<<<grid, 256, 0, stream>>>(xb, wqkvb, b_qkv, qkv,
                                                 Nn, QKV_COLS, D_EMB);
    }

    // 4) per-edge scores + segment max
    edge_scores<<<(E + 7) / 8, 256, 0, stream>>>(qkv, send, recv, scores, smaxkey, E);

    // 5) exp + segment sum
    {
        size_t total = (size_t)E * HEADS;
        edge_exp<<<(unsigned)((total + 255) / 256), 256, 0, stream>>>(
            scores, recv, smaxkey, denom, total);
    }

    // 6) weighted scatter of V
    edge_scatter<<<(E + 7) / 8, 256, 0, stream>>>(qkv, scores, denom, send, recv, msg, E);

    // 7) output projection: out = msg @ W_ff^T + b_ff   [Nn x 512]
    {
        size_t n = (size_t)Nn * D_EMB;
        cast_to_bf16<<<2048, 256, 0, stream>>>(msg, msgb, n);
        dim3 grid((Nn + 31) / 32, D_EMB / 512);
        wmma_gemm_bias<<<grid, 256, 0, stream>>>(msgb, wffb, b_ff, (float*)d_out,
                                                 Nn, D_EMB, D_EMB);
    }
}
